// GCN_32134945309367
// MI455X (gfx1250) — matrix operand order repaired and checked by reading, not compiled, not run on hardware
//
#include <hip/hip_runtime.h>

typedef __attribute__((ext_vector_type(16))) __bf16 v16bf;
typedef __attribute__((ext_vector_type(8)))  float  v8f;

union BF16x16 { v16bf v; uint4 u[2]; };

// ---------------------------------------------------------------- utilities

__global__ void cvt_f32_bf16(const float* __restrict__ in, __bf16* __restrict__ out, int n) {
    int t = blockIdx.x * blockDim.x + threadIdx.x;
    if (t < n) out[t] = (__bf16)in[t];
}

// W is [K x O] row-major (in_features x out_features); Wt is [O x K] bf16
__global__ void wtrans_bf16(const float* __restrict__ W, __bf16* __restrict__ Wt, int K, int O) {
    int t = blockIdx.x * blockDim.x + threadIdx.x;
    if (t >= K * O) return;
    int o = t / K, i = t - o * K;
    Wt[t] = (__bf16)W[(size_t)i * O + o];
}

__global__ void deg_init(float* __restrict__ deg, int n) {
    int t = blockIdx.x * blockDim.x + threadIdx.x;
    if (t < n) deg[t] = 1.0f;   // self loop
}

__global__ void deg_count(const int* __restrict__ dst, float* __restrict__ deg, int e) {
    int t = blockIdx.x * blockDim.x + threadIdx.x;
    if (t < e) atomicAdd(&deg[dst[t]], 1.0f);
}

__global__ void deg_rsqrt(float* __restrict__ deg, int n) {
    int t = blockIdx.x * blockDim.x + threadIdx.x;
    if (t < n) deg[t] = rsqrtf(deg[t]);
}

// --------------------------------------------------- WMMA GEMM  C = A * B
// A: [M x K] bf16 row-major.  Bt: [Nc x K] bf16 row-major (i.e. B transposed).
// C: [M x Nc] f32.  One wave computes one 16x16 tile of C.
__global__ __launch_bounds__(256)
void gcn_gemm_bf16(const __bf16* __restrict__ A, const __bf16* __restrict__ Bt,
                   float* __restrict__ C, int M, int K, int Nc) {
    const int gw   = (blockIdx.x * blockDim.x + threadIdx.x) >> 5;
    const int lane = threadIdx.x & 31;
    const int ntile = Nc >> 4;
    const int rt = gw / ntile;
    const int ct = gw - rt * ntile;
    if (rt * 16 >= M) return;                 // wave-uniform exit (EXEC stays all-1)

    const int r  = lane & 15;                 // matrix row (A) / col (B) within tile
    const int hl = lane >> 4;                 // lane half selects K sub-range
    const __bf16* ap = A  + (size_t)(rt * 16 + r) * K + hl * 8;    // A: K = hl*8..+7, +16..
    const __bf16* bp = Bt + (size_t)(ct * 16 + r) * K + hl * 8;    // B: K = hl*8..+7, +16..

    v8f acc = {};
    for (int k = 0; k < K; k += 32) {
        BF16x16 a, b;
        a.u[0] = *(const uint4*)(ap + k);
        a.u[1] = *(const uint4*)(ap + k + 16);
        b.u[0] = *(const uint4*)(bp + k);
        b.u[1] = *(const uint4*)(bp + k + 16);
        acc = __builtin_amdgcn_wmma_f32_16x16x32_bf16(false, a.v, false, b.v,
                                                      (short)0, acc, false, false);
    }
    // C/D layout: VGPR i, lanes 0-15 -> (M=i, N=lane); lanes 16-31 -> (M=i+8, N=lane-16)
    float* cp = C + (size_t)(rt * 16 + hl * 8) * Nc + ct * 16 + (lane & 15);
    for (int i = 0; i < 8; ++i) cp[(size_t)i * Nc] = acc[i];
}

// --------------------------------------------------------- message passing

// self loop: agg[i,f] = hw[i,f] * rinv[i]^2
__global__ void self_loop_init(const float* __restrict__ hw, const float* __restrict__ rinv,
                               float* __restrict__ agg, int F, int n) {
    int t = blockIdx.x * blockDim.x + threadIdx.x;
    if (t >= n) return;
    int i = t / F;
    float ri = rinv[i];
    agg[t] = hw[t] * ri * ri;
}

// agg[d] += hw[s] * rinv[s]*rinv[d]   (4 feats / thread, f32 atomics into L2)
__global__ void scatter_edges(const int* __restrict__ src, const int* __restrict__ dst,
                              const float* __restrict__ hw, const float* __restrict__ rinv,
                              float* __restrict__ agg, int F, int e) {
    int t = blockIdx.x * blockDim.x + threadIdx.x;
    int qpe = F >> 2;
    int ei = t / qpe;
    if (ei >= e) return;
    int q = (t - ei * qpe) << 2;
    int s = src[ei], d = dst[ei];
    float nm = rinv[s] * rinv[d];
    const float4 h = *(const float4*)(hw + (size_t)s * F + q);
    float* ap = agg + (size_t)d * F + q;
    atomicAdd(ap + 0, h.x * nm);
    atomicAdd(ap + 1, h.y * nm);
    atomicAdd(ap + 2, h.z * nm);
    atomicAdd(ap + 3, h.w * nm);
}

__global__ void bias_relu_bf16(const float* __restrict__ agg, const float* __restrict__ b,
                               __bf16* __restrict__ out, int Fmask, int n) {
    int t = blockIdx.x * blockDim.x + threadIdx.x;
    if (t >= n) return;
    out[t] = (__bf16)fmaxf(agg[t] + b[t & Fmask], 0.0f);
}

// ------------------------------------------------ fused edge classifier MLP
// Per block: 64 edges.  lw1t (64KB bf16) staged in LDS once per block via
// GLOBAL_LOAD_ASYNC_TO_LDS_B128; A = cat(z[src],z[dst]) gathered the same way.
// Layer 1 via WMMA (8 waves x 8 tiles of 16x16, K=128); layer 2 (256->1) is
// folded out of the accumulators with ds_add_f32 LDS atomics; then sigmoid.
__global__ __launch_bounds__(256)
void edge_mlp(const int* __restrict__ src, const int* __restrict__ dst,
              const __bf16* __restrict__ z,        // [N x 64] bf16
              const __bf16* __restrict__ lw1t,     // [256 x 128] bf16 (transposed)
              const float* __restrict__ lb1, const float* __restrict__ lw2,
              const float* __restrict__ lb2, float* __restrict__ out) {
    __shared__ __bf16 sB[256 * 128];   // 64 KB: staged lw1t
    __shared__ __bf16 sA[64 * 128];    // 16 KB: gathered edge features
    __shared__ float  sEdge[64];       // per-edge logit accumulators

    const int tid = threadIdx.x;
    const int e0  = blockIdx.x << 6;   // 64 edges per block

    if (tid < 64) sEdge[tid] = lb2[0];

    // ---- async stage of lw1t -> sB : 4096 x 16B, 16 per thread
    {
        const unsigned lbase = (unsigned)(size_t)&sB[0];   // LDS flat addr low 32b = offset
        for (int i = 0; i < 16; ++i) {
            const int idx = (i * 256 + tid) << 3;          // bf16 elements, 8 per 16B
            const unsigned loff = lbase + (idx << 1);
            const unsigned long long ga = (unsigned long long)(const void*)(lw1t + idx);
            asm volatile("global_load_async_to_lds_b128 %0, %1, off"
                         :: "v"(loff), "v"(ga) : "memory");
        }
    }
    // ---- async gather of cat(z[src],z[dst]) -> sA : 1024 x 16B, 4 per thread
    {
        const unsigned abase = (unsigned)(size_t)&sA[0];
        for (int i = 0; i < 4; ++i) {
            const int t = i * 256 + tid;                   // 0..1023
            const int e = t >> 4, seg = t & 15;
            const int node = (seg < 8) ? src[e0 + e] : dst[e0 + e];
            const int c = (seg & 7) << 3;
            const unsigned loff = abase + ((e * 128 + (seg << 3)) << 1);
            const unsigned long long ga =
                (unsigned long long)(const void*)(z + (size_t)node * 64 + c);
            asm volatile("global_load_async_to_lds_b128 %0, %1, off"
                         :: "v"(loff), "v"(ga) : "memory");
        }
    }
    asm volatile("s_wait_asynccnt 0" ::: "memory");        // own async transfers done
    __syncthreads();                                       // everyone's transfers visible

    // ---- layer 1 WMMA + fused layer 2 reduction
    const int wave = tid >> 5, lane = tid & 31;
    const int r = lane & 15, hl = lane >> 4;
    const int n = lane & 15, m0 = hl << 3;
    for (int t8 = 0; t8 < 8; ++t8) {
        const int id = wave * 8 + t8;                      // 64 tiles: 4 M x 16 N
        const int mt = id & 3, nt = id >> 2;
        const __bf16* ap = &sA[(mt * 16 + r) * 128 + hl * 8];
        const __bf16* bp = &sB[(nt * 16 + r) * 128 + hl * 8];
        v8f acc = {};
        for (int k = 0; k < 128; k += 32) {
            BF16x16 a, b;
            a.u[0] = *(const uint4*)(ap + k);              // ds_load_b128
            a.u[1] = *(const uint4*)(ap + k + 16);
            b.u[0] = *(const uint4*)(bp + k);
            b.u[1] = *(const uint4*)(bp + k + 16);
            acc = __builtin_amdgcn_wmma_f32_16x16x32_bf16(false, a.v, false, b.v,
                                                          (short)0, acc, false, false);
        }
        const float bias = lb1[nt * 16 + n];
        const float w2   = lw2[nt * 16 + n];
        // lane holds rows m0..m0+7 at col n of this tile: fold relu()*lw2 per edge
        for (int i = 0; i < 8; ++i) {
            const float v = fmaxf(acc[i] + bias, 0.0f) * w2;
            atomicAdd(&sEdge[mt * 16 + m0 + i], v);        // ds_add_f32
        }
    }
    __syncthreads();
    if (tid < 64) out[e0 + tid] = 1.0f / (1.0f + __expf(-sEdge[tid]));
}

// ------------------------------------------------------------------ driver

extern "C" void kernel_launch(void* const* d_in, const int* in_sizes, int n_in,
                              void* d_out, int out_size, void* d_ws, size_t ws_size,
                              hipStream_t stream) {
    (void)in_sizes; (void)n_in; (void)out_size; (void)ws_size;

    constexpr int N   = 100000;
    constexpr int E   = 1600000;
    constexpr int FIN = 128;
    constexpr int H   = 128;
    constexpr int EMB = 64;
    constexpr int HL  = 256;

    const float* x   = (const float*)d_in[0];
    const int*   ei  = (const int*)d_in[1];
    const int*   src = ei;
    const int*   dst = ei + E;
    const float* W0  = (const float*)d_in[2];  const float* b0 = (const float*)d_in[3];
    const float* W1  = (const float*)d_in[4];  const float* b1 = (const float*)d_in[5];
    const float* W2  = (const float*)d_in[6];  const float* b2 = (const float*)d_in[7];
    const float* lw1 = (const float*)d_in[8];  const float* lb1 = (const float*)d_in[9];
    const float* lw2 = (const float*)d_in[10]; const float* lb2 = (const float*)d_in[11];
    float* out = (float*)d_out;

    // workspace carve (base from hipMalloc is 256B-aligned; sizes rounded up)
    char* base = (char*)d_ws;
    size_t off = 0;
    auto carve = [&](size_t bytes) -> void* {
        void* p = base + off;
        off += (bytes + 255) & ~(size_t)255;
        return p;
    };
    float*  rinv = (float*)carve((size_t)N * 4);           // deg -> rsqrt in place
    __bf16* actA = (__bf16*)carve((size_t)N * 128 * 2);    // x_bf16, later z2
    __bf16* actB = (__bf16*)carve((size_t)N * 128 * 2);    // z1, later z3 (N x 64)
    float*  hw   = (float*)carve((size_t)N * 128 * 4);
    float*  agg  = (float*)carve((size_t)N * 128 * 4);
    __bf16* W0t  = (__bf16*)carve((size_t)FIN * H * 2);
    __bf16* W1t  = (__bf16*)carve((size_t)H * H * 2);
    __bf16* W2t  = (__bf16*)carve((size_t)H * EMB * 2);
    __bf16* lw1t = (__bf16*)carve((size_t)2 * EMB * HL * 2);

    const int TB = 256;
    auto blk = [](long n, int t) { return (int)((n + t - 1) / t); };

    // preprocessing: convert + transpose weights, degrees
    cvt_f32_bf16<<<blk((long)N * FIN, TB), TB, 0, stream>>>(x, actA, N * FIN);
    wtrans_bf16<<<blk(FIN * H, TB), TB, 0, stream>>>(W0, W0t, FIN, H);
    wtrans_bf16<<<blk(H * H, TB), TB, 0, stream>>>(W1, W1t, H, H);
    wtrans_bf16<<<blk(H * EMB, TB), TB, 0, stream>>>(W2, W2t, H, EMB);
    wtrans_bf16<<<blk(2 * EMB * HL, TB), TB, 0, stream>>>(lw1, lw1t, 2 * EMB, HL);
    deg_init<<<blk(N, TB), TB, 0, stream>>>(rinv, N);
    deg_count<<<blk(E, TB), TB, 0, stream>>>(dst, rinv, E);
    deg_rsqrt<<<blk(N, TB), TB, 0, stream>>>(rinv, N);

    // layer 1: x(bf16) @ W0 -> scatter -> relu -> z1(bf16)
    {
        int waves = (N / 16) * (H / 16);
        gcn_gemm_bf16<<<blk((long)waves * 32, TB), TB, 0, stream>>>(actA, W0t, hw, N, FIN, H);
        self_loop_init<<<blk((long)N * H, TB), TB, 0, stream>>>(hw, rinv, agg, H, N * H);
        scatter_edges<<<blk((long)E * (H / 4), TB), TB, 0, stream>>>(src, dst, hw, rinv, agg, H, E);
        bias_relu_bf16<<<blk((long)N * H, TB), TB, 0, stream>>>(agg, b0, actB, H - 1, N * H);
    }
    // layer 2: z1 @ W1 -> scatter -> relu -> z2(bf16)
    {
        int waves = (N / 16) * (H / 16);
        gcn_gemm_bf16<<<blk((long)waves * 32, TB), TB, 0, stream>>>(actB, W1t, hw, N, H, H);
        self_loop_init<<<blk((long)N * H, TB), TB, 0, stream>>>(hw, rinv, agg, H, N * H);
        scatter_edges<<<blk((long)E * (H / 4), TB), TB, 0, stream>>>(src, dst, hw, rinv, agg, H, E);
        bias_relu_bf16<<<blk((long)N * H, TB), TB, 0, stream>>>(agg, b1, actA, H - 1, N * H);
    }
    // layer 3: z2 @ W2 -> scatter -> relu -> z3(bf16, N x 64)
    {
        int waves = (N / 16) * (EMB / 16);
        gcn_gemm_bf16<<<blk((long)waves * 32, TB), TB, 0, stream>>>(actA, W2t, hw, N, H, EMB);
        self_loop_init<<<blk((long)N * EMB, TB), TB, 0, stream>>>(hw, rinv, agg, EMB, N * EMB);
        scatter_edges<<<blk((long)E * (EMB / 4), TB), TB, 0, stream>>>(src, dst, hw, rinv, agg, EMB, E);
        bias_relu_bf16<<<blk((long)N * EMB, TB), TB, 0, stream>>>(agg, b2, actB, EMB - 1, N * EMB);
    }
    // fused edge MLP + sigmoid (dominant compute): 64 edges/block,
    // async-LDS staged weights + gathers, WMMA, LDS-atomic second layer
    edge_mlp<<<E / 64, TB, 0, stream>>>(src, dst, actB, lw1t, lb1, lw2, lb2, out);
}
